// EncoderBlock_71073118814329
// MI455X (gfx1250) — matrix operand order repaired and checked by reading, not compiled, not run on hardware
//
#include <hip/hip_runtime.h>

#define Bq 4
#define Sq 2048
#define Dq 1024
#define Hq 16
#define DKq 64
#define DFFq 4096
#define ROWS (Bq*Sq)            // 8192

typedef __attribute__((ext_vector_type(16))) __bf16        v16bf;
typedef __attribute__((ext_vector_type(8)))  float         v8f;
typedef __attribute__((ext_vector_type(8)))  unsigned int  v8u;

// ---------- helpers ----------
__device__ __forceinline__ unsigned short f2bf(float f) {
  unsigned int u = __builtin_bit_cast(unsigned int, f);
  u += 0x7fffu + ((u >> 16) & 1u);           // RNE
  return (unsigned short)(u >> 16);
}
__device__ __forceinline__ unsigned int pack_bf2(float lo, float hi) {
#if __has_builtin(__builtin_amdgcn_cvt_pk_bf16_f32)
  typedef __attribute__((ext_vector_type(2))) __bf16 v2bf;
  v2bf r = __builtin_amdgcn_cvt_pk_bf16_f32(lo, hi);
  return __builtin_bit_cast(unsigned int, r);
#else
  return (unsigned int)f2bf(lo) | ((unsigned int)f2bf(hi) << 16);
#endif
}

// A-matrix (16-bit, 16x32) K index for pair p, lane-half h (ISA 7.12.2)
__device__ __forceinline__ int a_kidx(int p, int half) {
  return ((p < 4) ? (2 * p) : (16 + 2 * (p - 4))) + half * 8;
}

// Load A fragment (16x32 bf16) from row-major [16][ldk] bf16 at `base`
__device__ __forceinline__ v16bf load_afrag(const unsigned short* base, int ldk, int lane) {
  int m = lane & 15, half = lane >> 4;
  const unsigned short* rp = base + (size_t)m * ldk;
  v8u u;
#pragma unroll
  for (int p = 0; p < 8; ++p)
    u[p] = *(const unsigned int*)(rp + a_kidx(p, half));
  return __builtin_bit_cast(v16bf, u);
}

// Load B fragment (32x16 bf16) from K-transposed layout [n][ldk] bf16 at `base`
__device__ __forceinline__ v16bf load_bfrag(const unsigned short* base, int ldk, int lane) {
  int n = lane & 15, half = lane >> 4;
  const unsigned short* rp = base + (size_t)n * ldk;
  v8u u;
#pragma unroll
  for (int p = 0; p < 8; ++p)
    u[p] = *(const unsigned int*)(rp + a_kidx(p, half));
  return __builtin_bit_cast(v16bf, u);
}

// Build A fragment from fp32 row-major [16][ldk] (for P in attention)
__device__ __forceinline__ v16bf load_afrag_f32(const float* base, int ldk, int lane) {
  int m = lane & 15, half = lane >> 4;
  const float* rp = base + (size_t)m * ldk;
  v8u u;
#pragma unroll
  for (int p = 0; p < 8; ++p) {
    int k = a_kidx(p, half);
    u[p] = pack_bf2(rp[k], rp[k + 1]);
  }
  return __builtin_bit_cast(v16bf, u);
}

// max-reduce across 16-lane row groups using DPP row_xmask (VALU, no LDS)
template <int MASK>
__device__ __forceinline__ float xmax_step(float x) {
  int xi = __builtin_bit_cast(int, x);
  int yi = __builtin_amdgcn_update_dpp(0, xi, 0x160 | MASK, 0xf, 0xf, true);
  return fmaxf(x, __builtin_bit_cast(float, yi));
}
__device__ __forceinline__ float red16_max(float x) {
  x = xmax_step<1>(x);
  x = xmax_step<2>(x);
  x = xmax_step<4>(x);
  x = xmax_step<8>(x);
  return x;
}

// ---------- fp32 -> packed bf16 (row-major copy) ----------
__global__ __launch_bounds__(256) void cvt_f32_bf16(const float* __restrict__ src,
                                                    unsigned int* __restrict__ dst,
                                                    int nPairs) {
  int i = blockIdx.x * blockDim.x + threadIdx.x;
  int stride = gridDim.x * blockDim.x;
  for (; i < nPairs; i += stride)
    dst[i] = pack_bf2(src[2 * i], src[2 * i + 1]);
}

// ---------- fp32 W[K][N] -> bf16 Wt[N][K] (one-time weight transpose) ----------
__global__ __launch_bounds__(256) void cvt_wt_kernel(const float* __restrict__ W,
                                                     unsigned short* __restrict__ Wt,
                                                     int K, int N) {
  int i = blockIdx.x * blockDim.x + threadIdx.x;
  int stride = gridDim.x * blockDim.x;
  int total = K * N;
  for (; i < total; i += stride) {
    int k = i / N, n = i - k * N;            // reads coalesced over n
    Wt[(size_t)n * K + k] = f2bf(W[i]);
  }
}

// ---------- bf16 V[b*S][D] -> bf16 Vt[(b*H+h)*64+dk][S] ----------
__global__ __launch_bounds__(256) void vtrans_kernel(const unsigned short* __restrict__ vbuf,
                                                     unsigned short* __restrict__ vtb) {
  int i = blockIdx.x * blockDim.x + threadIdx.x;
  int stride = gridDim.x * blockDim.x;
  const int total = ROWS * Dq;
  for (; i < total; i += stride) {
    int row = i >> 10, col = i & 1023;       // row=(b,s), col=(h,dk)
    int b = row >> 11, s = row & 2047;
    int h = col >> 6, dk = col & 63;
    vtb[(size_t)(((b * Hq + h) << 6) + dk) * Sq + s] = vbuf[i];
  }
}

// ---------- GEMM: C[M,N] = A[M,K](bf16) @ Wt[N,K](bf16, pre-transposed) + bias ----
// block tile 128x128, 8 waves as 4x2 of 32x64 wave tiles, K-step 64,
// register-staged global->LDS prefetch pipeline
template <int RELU, int OUTBF>
__global__ __launch_bounds__(256) void gemm_tn_kernel(
    const unsigned short* __restrict__ A, const unsigned short* __restrict__ Wt,
    const float* __restrict__ bias, float* __restrict__ Cf,
    unsigned short* __restrict__ Cb, int M, int N, int K) {
  __shared__ unsigned short Asm[128 * 64];
  __shared__ unsigned short Wsm[128 * 64];   // [n][k]

  const int tid  = threadIdx.x;
  const int lane = tid & 31;
  const int wave = tid >> 5;
  const int waveM = wave & 3, waveN = wave >> 2;
  const int half = lane >> 4;
  const int keyn = lane & 15;
  const int blkM = blockIdx.y * 128;
  const int blkN = blockIdx.x * 128;

  v8f acc[2][4];
#pragma unroll
  for (int i = 0; i < 2; ++i)
#pragma unroll
    for (int j = 0; j < 4; ++j) acc[i][j] = {};

  uint4 ra[4], rw[4];
  auto gload = [&](int k0) {
#pragma unroll
    for (int i = 0; i < 4; ++i) {
      int idx = tid + i * 256;
      int row = idx >> 3, c8 = idx & 7;
      ra[i] = *(const uint4*)&A [(size_t)(blkM + row) * K + k0 + c8 * 8];
      rw[i] = *(const uint4*)&Wt[(size_t)(blkN + row) * K + k0 + c8 * 8];
    }
  };

  gload(0);
  for (int k0 = 0; k0 < K; k0 += 64) {
    __syncthreads();                          // prior compute finished reading LDS
#pragma unroll
    for (int i = 0; i < 4; ++i) {
      int idx = tid + i * 256;
      int row = idx >> 3, c8 = idx & 7;
      *(uint4*)&Asm[row * 64 + c8 * 8] = ra[i];
      *(uint4*)&Wsm[row * 64 + c8 * 8] = rw[i];
    }
    __syncthreads();                          // tile ready
    if (k0 + 64 < K) gload(k0 + 64);          // overlap next loads with WMMAs

#pragma unroll
    for (int kc = 0; kc < 2; ++kc) {
      v16bf a0 = load_afrag(&Asm[(waveM * 32 + 0)  * 64 + kc * 32], 64, lane);
      v16bf a1 = load_afrag(&Asm[(waveM * 32 + 16) * 64 + kc * 32], 64, lane);
#pragma unroll
      for (int tn = 0; tn < 4; ++tn) {
        v16bf bf = load_bfrag(&Wsm[(waveN * 64 + tn * 16) * 64 + kc * 32], 64, lane);
        acc[0][tn] = __builtin_amdgcn_wmma_f32_16x16x32_bf16(false, a0, false, bf, (short)0, acc[0][tn], false, false);
        acc[1][tn] = __builtin_amdgcn_wmma_f32_16x16x32_bf16(false, a1, false, bf, (short)0, acc[1][tn], false, false);
      }
    }
  }

#pragma unroll
  for (int tn = 0; tn < 4; ++tn) {
    int n = blkN + waveN * 64 + tn * 16 + keyn;
    float bv = bias[n];
#pragma unroll
    for (int tm = 0; tm < 2; ++tm)
#pragma unroll
      for (int r = 0; r < 8; ++r) {
        int m = blkM + waveM * 32 + tm * 16 + half * 8 + r;
        float v = acc[tm][tn][r] + bv;
        if (RELU) v = fmaxf(v, 0.0f);
        if (OUTBF) Cb[(size_t)m * N + n] = f2bf(v);
        else       Cf[(size_t)m * N + n] = v;
      }
  }
}

// ---------- flash attention ----------
// grid (S/128, B*H); 8 waves; wave owns 16 q rows; 32-key chunks, double-buffered
__global__ __launch_bounds__(256) void attn_kernel(
    const unsigned short* __restrict__ qb, const unsigned short* __restrict__ kb,
    const unsigned short* __restrict__ vtb, const int* __restrict__ mask,
    unsigned short* __restrict__ ctxb) {
  __shared__ unsigned short Ksm[2][32 * 64];   // [key][dk]
  __shared__ unsigned short Vtsm[2][64 * 32];  // [dk][key]
  __shared__ float Psm[8 * 16 * 32];           // per-wave P staging

  const int tid  = threadIdx.x;
  const int lane = tid & 31;
  const int wave = tid >> 5;
  const int half = lane >> 4;
  const int keyn = lane & 15;
  const int q0   = blockIdx.x * 128;
  const int bh   = blockIdx.y;
  const int b    = bh / Hq, h = bh % Hq;
  const float scale = 0.125f; // 1/sqrt(64)

  // Q fragments (registers for whole loop)
  const unsigned short* qbase =
      qb + (size_t)(b * Sq + q0 + wave * 16) * Dq + h * DKq;
  v16bf qf0 = load_afrag(qbase, Dq, lane);
  v16bf qf1 = load_afrag(qbase + 32, Dq, lane);

  // ones B-fragment for row-sum WMMA
  v8u ouv;
#pragma unroll
  for (int p = 0; p < 8; ++p) ouv[p] = 0x3f803f80u;
  const v16bf onesB = __builtin_bit_cast(v16bf, ouv);

  float rm[8], rl[8];
  v8f o[4];
#pragma unroll
  for (int r = 0; r < 8; ++r) { rm[r] = -1e30f; rl[r] = 0.0f; }
#pragma unroll
  for (int t = 0; t < 4; ++t) o[t] = {};

  float* Pw = &Psm[wave * 512];

  const unsigned short* krow = kb + (size_t)(b * Sq) * Dq + h * DKq;
  const unsigned short* vrow = vtb + (size_t)((b * Hq + h) << 6) * Sq;

  // stage chunk kc into buffer `buf`
  auto stage = [&](int buf, int kc) {
    int key = tid >> 3, c8 = tid & 7;
    *(uint4*)&Ksm[buf][key * 64 + c8 * 8] =
        *(const uint4*)&krow[(size_t)(kc + key) * Dq + c8 * 8];
    int d = tid >> 2, c4 = tid & 3;
    *(uint4*)&Vtsm[buf][d * 32 + c4 * 8] =
        *(const uint4*)&vrow[(size_t)d * Sq + kc + c4 * 8];
  };

  stage(0, 0);
  int cur = 0;
  for (int kc = 0; kc < Sq; kc += 32, cur ^= 1) {
    __syncthreads();
    if (kc + 32 < Sq) stage(cur ^ 1, kc + 32);

    // scores S[16,32] = Q(16x64) K^T(64x32)
    v8f s0 = {}, s1 = {};
    {
      v16bf kb0 = load_bfrag(&Ksm[cur][0 * 64 + 0],   64, lane);
      v16bf kb1 = load_bfrag(&Ksm[cur][16 * 64 + 0],  64, lane);
      s0 = __builtin_amdgcn_wmma_f32_16x16x32_bf16(false, qf0, false, kb0, (short)0, s0, false, false);
      s1 = __builtin_amdgcn_wmma_f32_16x16x32_bf16(false, qf0, false, kb1, (short)0, s1, false, false);
      v16bf kb2 = load_bfrag(&Ksm[cur][0 * 64 + 32],  64, lane);
      v16bf kb3 = load_bfrag(&Ksm[cur][16 * 64 + 32], 64, lane);
      s0 = __builtin_amdgcn_wmma_f32_16x16x32_bf16(false, qf1, false, kb2, (short)0, s0, false, false);
      s1 = __builtin_amdgcn_wmma_f32_16x16x32_bf16(false, qf1, false, kb3, (short)0, s1, false, false);
    }

    // additive mask bias: 2 selects per chunk instead of 16
    float ab0 = mask[b * Sq + kc + keyn]      ? 0.0f : -1e9f;
    float ab1 = mask[b * Sq + kc + 16 + keyn] ? 0.0f : -1e9f;

    float fac[8];
#pragma unroll
    for (int r = 0; r < 8; ++r) {
      float sc0 = fmaf(s0[r], scale, ab0);
      float sc1 = fmaf(s1[r], scale, ab1);
      float mx = red16_max(fmaxf(sc0, sc1));
      float mnew = fmaxf(rm[r], mx);
      fac[r] = __expf(rm[r] - mnew);
      rm[r] = mnew;
      float p0 = __expf(sc0 - mnew);
      float p1 = __expf(sc1 - mnew);
      int mrow = half * 8 + r;
      Pw[mrow * 32 + keyn]      = p0;
      Pw[mrow * 32 + 16 + keyn] = p1;
#pragma unroll
      for (int t = 0; t < 4; ++t) o[t][r] *= fac[r];
    }

    // P fragment (bf16) from LDS; row sums via P x ones WMMA
    v16bf pa = load_afrag_f32(Pw, 32, lane);
    v8f zero = {};
    v8f rs = __builtin_amdgcn_wmma_f32_16x16x32_bf16(false, pa, false, onesB, (short)0, zero, false, false);
#pragma unroll
    for (int r = 0; r < 8; ++r) rl[r] = rl[r] * fac[r] + rs[r];

    // O += P(16x32) V(32x64)
#pragma unroll
    for (int t = 0; t < 4; ++t) {
      v16bf vf = load_bfrag(&Vtsm[cur][(t * 16) * 32], 32, lane);
      o[t] = __builtin_amdgcn_wmma_f32_16x16x32_bf16(false, pa, false, vf, (short)0, o[t], false, false);
    }
  }

  // write context (bf16) at [b, q, h*64 + dk]
#pragma unroll
  for (int r = 0; r < 8; ++r) {
    float inv = 1.0f / rl[r];
    int m = q0 + wave * 16 + half * 8 + r;
#pragma unroll
    for (int t = 0; t < 4; ++t) {
      int n = h * DKq + t * 16 + keyn;
      ctxb[(size_t)(b * Sq + m) * Dq + n] = f2bf(o[t][r] * inv);
    }
  }
}

// ---------- residual + LayerNorm (ddof=1 std, eps AFTER division) ----------
__global__ __launch_bounds__(256) void ln_kernel(
    const float* __restrict__ resid, const float* __restrict__ delta,
    const float* __restrict__ g, const float* __restrict__ bt,
    float* __restrict__ outF, unsigned short* __restrict__ outB) {
  __shared__ float sred[256];
  const int row = blockIdx.x;
  const int tid = threadIdx.x;
  const float4* r4 = (const float4*)(resid + (size_t)row * Dq);
  const float4* d4 = (const float4*)(delta + (size_t)row * Dq);

  float4 ra = r4[tid], da = d4[tid];
  float t[4] = { ra.x + da.x, ra.y + da.y, ra.z + da.z, ra.w + da.w };
  float s = (t[0] + t[1]) + (t[2] + t[3]);

  sred[tid] = s; __syncthreads();
  for (int off = 128; off > 0; off >>= 1) {
    if (tid < off) sred[tid] += sred[tid + off];
    __syncthreads();
  }
  float mean = sred[0] * (1.0f / (float)Dq);
  __syncthreads();

  float ss = 0.0f;
#pragma unroll
  for (int i = 0; i < 4; ++i) { float d = t[i] - mean; ss += d * d; }
  sred[tid] = ss; __syncthreads();
  for (int off = 128; off > 0; off >>= 1) {
    if (tid < off) sred[tid] += sred[tid + off];
    __syncthreads();
  }
  float stdv = sqrtf(sred[0] * (1.0f / (float)(Dq - 1)));  // unbiased
  float gg = g[0], bb = bt[0];

  float y[4];
#pragma unroll
  for (int i = 0; i < 4; ++i) y[i] = gg * ((t[i] - mean) / stdv + 1e-9f) + bb;

  float4* o4 = (float4*)(outF + (size_t)row * Dq);
  o4[tid] = make_float4(y[0], y[1], y[2], y[3]);
  if (outB) {
    uint2 pb = make_uint2(pack_bf2(y[0], y[1]), pack_bf2(y[2], y[3]));
    ((uint2*)(outB + (size_t)row * Dq))[tid] = pb;
  }
}

// ---------- host ----------
extern "C" void kernel_launch(void* const* d_in, const int* in_sizes, int n_in,
                              void* d_out, int out_size, void* d_ws, size_t ws_size,
                              hipStream_t stream) {
  const float* x    = (const float*)d_in[0];
  const int*   mask = (const int*)d_in[1];
  const float* wq = (const float*)d_in[2];  const float* bq = (const float*)d_in[3];
  const float* wk = (const float*)d_in[4];  const float* bk = (const float*)d_in[5];
  const float* wv = (const float*)d_in[6];  const float* bv = (const float*)d_in[7];
  const float* wo = (const float*)d_in[8];  const float* bo = (const float*)d_in[9];
  const float* w1 = (const float*)d_in[10]; const float* b1 = (const float*)d_in[11];
  const float* w2 = (const float*)d_in[12]; const float* b2 = (const float*)d_in[13];
  const float* g1 = (const float*)d_in[14]; const float* bt1 = (const float*)d_in[15];
  const float* g2 = (const float*)d_in[16]; const float* bt2 = (const float*)d_in[17];
  float* out = (float*)d_out;

  char* w = (char*)d_ws;
  auto alloc = [&](size_t bytes) -> char* {
    char* p = w;
    w += (bytes + 255) & ~(size_t)255;
    return p;
  };
  unsigned short* xb   = (unsigned short*)alloc((size_t)ROWS * Dq * 2);
  unsigned short* wqt  = (unsigned short*)alloc((size_t)Dq * Dq * 2);
  unsigned short* wkt  = (unsigned short*)alloc((size_t)Dq * Dq * 2);
  unsigned short* wvt  = (unsigned short*)alloc((size_t)Dq * Dq * 2);
  unsigned short* wot  = (unsigned short*)alloc((size_t)Dq * Dq * 2);
  unsigned short* w1t  = (unsigned short*)alloc((size_t)Dq * DFFq * 2);
  unsigned short* w2t  = (unsigned short*)alloc((size_t)DFFq * Dq * 2);
  unsigned short* qbuf = (unsigned short*)alloc((size_t)ROWS * Dq * 2);
  unsigned short* kbuf = (unsigned short*)alloc((size_t)ROWS * Dq * 2);
  unsigned short* vbuf = (unsigned short*)alloc((size_t)ROWS * Dq * 2);
  unsigned short* vtb  = (unsigned short*)alloc((size_t)ROWS * Dq * 2);
  unsigned short* ctxb = (unsigned short*)alloc((size_t)ROWS * Dq * 2);
  float* attn_out      = (float*)alloc((size_t)ROWS * Dq * 4);
  float* x1f           = (float*)alloc((size_t)ROWS * Dq * 4);
  unsigned short* x1b  = (unsigned short*)alloc((size_t)ROWS * Dq * 2);
  unsigned short* hb   = (unsigned short*)alloc((size_t)ROWS * DFFq * 2);
  float* ff            = (float*)alloc((size_t)ROWS * Dq * 4);
  (void)ws_size; (void)n_in; (void)in_sizes; (void)out_size;

  // x -> bf16 (row-major)
  {
    int pairs = (int)((size_t)ROWS * Dq / 2);
    int blocks = (pairs + 255) / 256; if (blocks > 4096) blocks = 4096;
    cvt_f32_bf16<<<blocks, 256, 0, stream>>>(x, (unsigned int*)xb, pairs);
  }
  // weights -> bf16 transposed [N][K]
  auto cvtw = [&](const float* s, unsigned short* d, int K, int N) {
    int total = K * N;
    int blocks = (total + 255) / 256; if (blocks > 8192) blocks = 8192;
    cvt_wt_kernel<<<blocks, 256, 0, stream>>>(s, d, K, N);
  };
  cvtw(wq, wqt, Dq, Dq);
  cvtw(wk, wkt, Dq, Dq);
  cvtw(wv, wvt, Dq, Dq);
  cvtw(wo, wot, Dq, Dq);
  cvtw(w1, w1t, Dq, DFFq);
  cvtw(w2, w2t, DFFq, Dq);

  // Q, K, V projections (bf16 out)
  {
    dim3 g(Dq / 128, ROWS / 128);
    gemm_tn_kernel<0, 1><<<g, 256, 0, stream>>>(xb, wqt, bq, nullptr, qbuf, ROWS, Dq, Dq);
    gemm_tn_kernel<0, 1><<<g, 256, 0, stream>>>(xb, wkt, bk, nullptr, kbuf, ROWS, Dq, Dq);
    gemm_tn_kernel<0, 1><<<g, 256, 0, stream>>>(xb, wvt, bv, nullptr, vbuf, ROWS, Dq, Dq);
  }
  // V -> per-head transposed layout for attention B-fragments
  {
    int total = ROWS * Dq;
    int blocks = (total + 255) / 256; if (blocks > 8192) blocks = 8192;
    vtrans_kernel<<<blocks, 256, 0, stream>>>(vbuf, vtb);
  }
  // attention -> ctx (bf16)
  {
    dim3 g(Sq / 128, Bq * Hq);
    attn_kernel<<<g, 256, 0, stream>>>(qbuf, kbuf, vtb, mask, ctxb);
  }
  // output projection (f32 out)
  {
    dim3 g(Dq / 128, ROWS / 128);
    gemm_tn_kernel<0, 0><<<g, 256, 0, stream>>>(ctxb, wot, bo, attn_out, nullptr, ROWS, Dq, Dq);
  }
  // residual + LN1 -> x1 (f32 + bf16)
  ln_kernel<<<ROWS, 256, 0, stream>>>(x, attn_out, g1, bt1, x1f, x1b);
  // FFN up with ReLU (bf16 out)
  {
    dim3 g(DFFq / 128, ROWS / 128);
    gemm_tn_kernel<1, 1><<<g, 256, 0, stream>>>(x1b, w1t, b1, nullptr, hb, ROWS, DFFq, Dq);
  }
  // FFN down (f32 out)
  {
    dim3 g(Dq / 128, ROWS / 128);
    gemm_tn_kernel<0, 0><<<g, 256, 0, stream>>>(hb, w2t, b2, ff, nullptr, ROWS, Dq, DFFq);
  }
  // residual + LN2 -> d_out
  ln_kernel<<<ROWS, 256, 0, stream>>>(x1f, ff, g2, bt2, out, nullptr);
}
